// RGNN_6004364280394
// MI455X (gfx1250) — matrix operand order repaired and checked by reading, not compiled, not run on hardware
//
#include <hip/hip_runtime.h>
#include <hip/hip_bf16.h>
#include <stdint.h>

typedef __attribute__((ext_vector_type(16))) _Float16 v16h;
typedef __attribute__((ext_vector_type(8)))  float    v8f;
typedef int v4i_b128 __attribute__((vector_size(16)));
typedef __attribute__((address_space(1))) v4i_b128* gptr_v4i;
typedef __attribute__((address_space(3))) v4i_b128* lptr_v4i;

#define HD        256   // heads * dim = 4*64
#define DH        64
#define NEGSLOPE  0.2f

// ---------- gfx1250 async global->LDS (guarded: falls back to plain LDS stores) ----------
#if __has_builtin(__builtin_amdgcn_global_load_async_to_lds_b128)
#define USE_ASYNC_LDS 1
__device__ __forceinline__ void gld_async16(const _Float16* g, _Float16* l) {
  __builtin_amdgcn_global_load_async_to_lds_b128(
      (gptr_v4i)(uintptr_t)g,
      (lptr_v4i)(uint32_t)(uintptr_t)l,
      0, 0);
}
__device__ __forceinline__ void wait_async0() {
#if __has_builtin(__builtin_amdgcn_s_wait_asynccnt)
  __builtin_amdgcn_s_wait_asynccnt(0);
#else
  asm volatile("s_wait_asynccnt 0x0" ::: "memory");
#endif
}
#endif

// ---------- helpers ----------
__device__ __forceinline__ int fenc(float f) {           // order-preserving float->int
  int i = __float_as_int(f);
  return i ^ ((i >> 31) & 0x7fffffff);
}
__device__ __forceinline__ float fdec(int i) {
  return __int_as_float(i ^ ((i >> 31) & 0x7fffffff));
}
__device__ __forceinline__ float lrelu(float v) { return v > 0.f ? v : NEGSLOPE * v; }

// ---------- fills / converts ----------
__global__ void fill_f32(float* __restrict__ p, float v, long n) {
  long i = (long)blockIdx.x * blockDim.x + threadIdx.x;
  if (i < n) p[i] = v;
}
__global__ void fill_i32(int* __restrict__ p, int v, long n) {
  long i = (long)blockIdx.x * blockDim.x + threadIdx.x;
  if (i < n) p[i] = v;
}
__global__ void cvt_f16(const float* __restrict__ in, _Float16* __restrict__ out, long n) {
  long i = (long)blockIdx.x * blockDim.x + threadIdx.x;
  if (i < n) out[i] = (_Float16)in[i];
}
__global__ void relu_cvt(float* __restrict__ p, _Float16* __restrict__ o, long n) {
  long i = (long)blockIdx.x * blockDim.x + threadIdx.x;
  if (i < n) { float v = fmaxf(p[i], 0.f); p[i] = v; o[i] = (_Float16)v; }
}

// ---------- WMMA GEMM: C[M,N] = A16[M,K] * B32[K,N] (+bias[N]) ----------
// Block = 256 threads = 8 waves; block tile 64x64, BK=64 (2 WMMA k-steps / stage).
// A (f16, byte-copyable) staged via async-to-LDS when available; B (f32 weights)
// converted to f16 and transposed into [n][k] during staging.
__global__ __launch_bounds__(256) void gemm_wmma_f16(
    const _Float16* __restrict__ A, const float* __restrict__ B,
    float* __restrict__ C, const float* __restrict__ bias,
    int M, int N, int K)
{
  __shared__ _Float16 Al[64][72];   // [m][k], +8 halves row pad (rows stay 16B-aligned)
  __shared__ _Float16 Bl[64][72];   // [n][k] transposed tile

  const int tid  = threadIdx.x;
  const int wave = tid >> 5, lane = tid & 31;
  const int lm   = lane & 15, g = lane >> 4;
  const int m0   = blockIdx.y * 64, n0 = blockIdx.x * 64;
  const int wm   = (wave >> 1) << 4;   // 0,16,32,48
  const int wn   = (wave &  1) << 5;   // 0,32

  v8f acc0 = {}; v8f acc1 = {};

  // A loader: 512 16B-chunks per stage, 2 per thread
  const int ca0 = tid * 2, ca1 = tid * 2 + 1;
  const int ar0_ = ca0 >> 3, ao0 = (ca0 & 7) * 8;
  const int ar1_ = ca1 >> 3, ao1 = (ca1 & 7) * 8;
  // B loader: thread t handles row k=t>>2, 16 cols starting at (t&3)*16
  const int bk8 = tid >> 2, bc8 = (tid & 3) * 16;

  for (int kb = 0; kb < K; kb += 64) {
    // ---- stage A (f16 global -> f16 LDS, byte copy; OOB rows skipped, never stored) ----
    {
      const _Float16* s0 = A + (size_t)(m0 + ar0_) * K + kb + ao0;
      const _Float16* s1 = A + (size_t)(m0 + ar1_) * K + kb + ao1;
      _Float16* d0 = &Al[ar0_][ao0];
      _Float16* d1 = &Al[ar1_][ao1];
#if defined(USE_ASYNC_LDS)
      if (m0 + ar0_ < M) gld_async16(s0, d0);
      if (m0 + ar1_ < M) gld_async16(s1, d1);
#else
      if (m0 + ar0_ < M) *(uint4*)d0 = *(const uint4*)s0;
      if (m0 + ar1_ < M) *(uint4*)d1 = *(const uint4*)s1;
#endif
    }
    // ---- stage B (f32 global -> f16 LDS, transpose into [n][k]) ----
    {
      const float* p = B + (size_t)(kb + bk8) * N + n0 + bc8;
      float4 f0 = *(const float4*)p,       f1 = *(const float4*)(p + 4);
      float4 f2 = *(const float4*)(p + 8), f3 = *(const float4*)(p + 12);
      _Float16 h[16] = {
        (_Float16)f0.x, (_Float16)f0.y, (_Float16)f0.z, (_Float16)f0.w,
        (_Float16)f1.x, (_Float16)f1.y, (_Float16)f1.z, (_Float16)f1.w,
        (_Float16)f2.x, (_Float16)f2.y, (_Float16)f2.z, (_Float16)f2.w,
        (_Float16)f3.x, (_Float16)f3.y, (_Float16)f3.z, (_Float16)f3.w };
#pragma unroll
      for (int i = 0; i < 16; ++i) Bl[bc8 + i][bk8] = h[i];
    }
#if defined(USE_ASYNC_LDS)
    wait_async0();
#endif
    __syncthreads();

#pragma unroll
    for (int ks = 0; ks < 64; ks += 32) {
      union { v16h v; unsigned u[8]; } af, bf0, bf1;
      const _Float16* arow  = &Al[wm + lm][ks];
      const _Float16* bcol0 = &Bl[wn + lm][ks];
      const _Float16* bcol1 = &Bl[wn + 16 + lm][ks];
#pragma unroll
      for (int j = 0; j < 8; ++j) {
        // A 16x32 f16, this code's contiguous k order (same as B): lane group g holds K = 16g+2j,+1 at row lm
        int ka = 16 * g + 2 * j;
        // B 32x16 f16, this code's contiguous k order (same as A): lane group g holds K = 16g+2j,+1 at column lm
        int kq = 16 * g + 2 * j;
        af.u[j]  = *(const unsigned*)(arow  + ka);
        bf0.u[j] = *(const unsigned*)(bcol0 + kq);
        bf1.u[j] = *(const unsigned*)(bcol1 + kq);
      }
      acc0 = __builtin_amdgcn_wmma_f32_16x16x32_f16(false, af.v, false, bf0.v,
                                                    (short)0, acc0, false, false);
      acc1 = __builtin_amdgcn_wmma_f32_16x16x32_f16(false, af.v, false, bf1.v,
                                                    (short)0, acc1, false, false);
    }
    __syncthreads();
  }

  // C layout: VGPR r holds element (M = r + 8g, N = lm)
#pragma unroll
  for (int r = 0; r < 8; ++r) {
    int row = m0 + wm + r + 8 * g;
    if (row < M) {
      int c0 = n0 + wn + lm, c1 = c0 + 16;
      float v0 = acc0[r], v1 = acc1[r];
      if (bias) { v0 += bias[c0]; v1 += bias[c1]; }
      C[(size_t)row * N + c0] = v0;
      C[(size_t)row * N + c1] = v1;
    }
  }
}

// ---------- el[n,h] = sum_d fs[n,h,d]*al[h,d] ----------
__global__ void el_reduce(const float* __restrict__ fs, const float* __restrict__ al,
                          float* __restrict__ el, int Nn)
{
  int idx = blockIdx.x * blockDim.x + threadIdx.x;
  if (idx >= Nn * 4) return;
  int n = idx >> 2, h = idx & 3;
  const float* f = fs + (size_t)n * HD + h * DH;
  const float* a = al + h * DH;
  float sum = 0.f;
#pragma unroll
  for (int d = 0; d < DH; d += 4) {
    float4 fv = *(const float4*)(f + d);
    float4 av = *(const float4*)(a + d);
    sum += fv.x*av.x + fv.y*av.y + fv.z*av.z + fv.w*av.w;
  }
  el[idx] = sum;
}

// ---------- war[k,h] = sum_d W[k,h*64+d]*ar[h,d]  (folds the er GEMM into a GEMV) ----------
__global__ void w_ar_reduce(const float* __restrict__ W, const float* __restrict__ ar,
                            float* __restrict__ war, int K)
{
  int idx = blockIdx.x * blockDim.x + threadIdx.x;
  if (idx >= K * 4) return;
  int k = idx >> 2, h = idx & 3;
  const float* wr = W + (size_t)k * HD + h * DH;
  const float* a  = ar + h * DH;
  float sum = 0.f;
#pragma unroll
  for (int d = 0; d < DH; ++d) sum += wr[d] * a[d];
  war[idx] = sum;
}

// ---------- er[n,h] = sum_k X[n,k]*war[k,h] ----------
__global__ __launch_bounds__(256) void er_reduce(const float* __restrict__ X,
                                                 const float* __restrict__ war,
                                                 float* __restrict__ er, int Nd, int K)
{
  __shared__ float ws[1024];
  for (int i = threadIdx.x; i < K * 4; i += 256) ws[i] = war[i];
  __syncthreads();
  int idx = blockIdx.x * 256 + threadIdx.x;
  if (idx >= Nd * 4) return;
  int n = idx >> 2, h = idx & 3;
  const float* x = X + (size_t)n * K;
  float sum = 0.f;
  for (int k = 0; k < K; ++k) sum += x[k] * ws[k * 4 + h];
  er[idx] = sum;
}

// ---------- pass 1: per-dst max of leaky(el[src]+er[dst]) via ordered-int atomicMax ----------
__global__ void edge_max(const int* __restrict__ src, const int* __restrict__ dst,
                         const float* __restrict__ el, const float* __restrict__ er,
                         int* __restrict__ m, int E)
{
  int e = blockIdx.x * blockDim.x + threadIdx.x;
  if (e >= E) return;
  int sN = src[e], dN = dst[e];
  float4 a = *(const float4*)(el + (size_t)sN * 4);
  float4 b = *(const float4*)(er + (size_t)dN * 4);
  atomicMax(m + (size_t)dN * 4 + 0, fenc(lrelu(a.x + b.x)));
  atomicMax(m + (size_t)dN * 4 + 1, fenc(lrelu(a.y + b.y)));
  atomicMax(m + (size_t)dN * 4 + 2, fenc(lrelu(a.z + b.z)));
  atomicMax(m + (size_t)dN * 4 + 3, fenc(lrelu(a.w + b.w)));
}

// ---------- pass 2 (fused): ex = exp(e-m); s[dst]+=ex; acc[dst]+=ex*fs[src] ----------
__global__ __launch_bounds__(256) void edge_acc(
    const int* __restrict__ src, const int* __restrict__ dst,
    const float* __restrict__ el, const float* __restrict__ er,
    const int* __restrict__ m, const float* __restrict__ fs,
    float* __restrict__ s, float* __restrict__ acc, int E)
{
  int e = blockIdx.x * 8 + (threadIdx.x >> 5);
  int lane = threadIdx.x & 31;
  if (e >= E) return;
  int sN = src[e], dN = dst[e];
  float4 a = *(const float4*)(el + (size_t)sN * 4);
  float4 b = *(const float4*)(er + (size_t)dN * 4);
  const int* mp = m + (size_t)dN * 4;
  float exh[4];
  exh[0] = __expf(lrelu(a.x + b.x) - fdec(mp[0]));
  exh[1] = __expf(lrelu(a.y + b.y) - fdec(mp[1]));
  exh[2] = __expf(lrelu(a.z + b.z) - fdec(mp[2]));
  exh[3] = __expf(lrelu(a.w + b.w) - fdec(mp[3]));
  if (lane < 4) atomicAdd(s + (size_t)dN * 4 + lane, exh[lane]);
  const float* f = fs  + (size_t)sN * HD;
  float*       o = acc + (size_t)dN * HD;
  __builtin_prefetch(f + lane * 8, 0, 0);  // global_prefetch_b8
#pragma unroll
  for (int i = 0; i < 8; ++i) {
    int c = i * 32 + lane;
    atomicAdd(o + c, exh[c >> 6] * f[c]);
  }
}

// ---------- O[n,c] += (s>0 ? acc/s : 0) + b[c] ----------
__global__ void finalize_acc(const float* __restrict__ acc, const float* __restrict__ s,
                             const float* __restrict__ b, float* __restrict__ O, int Nd)
{
  long idx = (long)blockIdx.x * 256 + threadIdx.x;
  if (idx >= (long)Nd * HD) return;
  int  c = (int)(idx & 255), h = c >> 6;
  long n = idx >> 8;
  float sv = s[n * 4 + h];
  float v  = (sv > 0.f) ? acc[idx] / sv : 0.f;
  O[idx] += v + b[c];
}

// ---------- host orchestration ----------
static inline unsigned gblocks(long n, int b) { return (unsigned)((n + b - 1) / b); }

extern "C" void kernel_launch(void* const* d_in, const int* in_sizes, int n_in,
                              void* d_out, int out_size, void* d_ws, size_t ws_size,
                              hipStream_t stream)
{
  const float* x_paper  = (const float*)d_in[0];
  const float* x_author = (const float*)d_in[1];
  const float* W0   = (const float*)d_in[2];   // [3,128,256]
  const float* al0  = (const float*)d_in[3];   // [3,4,64]
  const float* ar0  = (const float*)d_in[4];
  const float* b0   = (const float*)d_in[5];   // [3,256]
  const float* W1   = (const float*)d_in[6];   // [3,256,256]
  const float* al1  = (const float*)d_in[7];
  const float* ar1  = (const float*)d_in[8];
  const float* b1   = (const float*)d_in[9];
  const float* lw   = (const float*)d_in[10];  // [256,192]
  const float* lb   = (const float*)d_in[11];  // [192]
  const int* src_c  = (const int*)d_in[12]; const int* dst_c  = (const int*)d_in[13];
  const int* src_w  = (const int*)d_in[14]; const int* dst_w  = (const int*)d_in[15];
  const int* src_wr = (const int*)d_in[16]; const int* dst_wr = (const int*)d_in[17];

  const int Np = in_sizes[0] / 128;
  const int Na = in_sizes[1] / 128;
  const int Ec = in_sizes[12], Ew = in_sizes[14], Ewr = in_sizes[16];

  // ---- workspace carve ----
  float* w = (float*)d_ws;
  const size_t P = (size_t)Np * HD, Aa = (size_t)Na * HD;
  float* p0  = w; w += P;      // layer0 paper out / layer1 input (f32)
  float* a0  = w; w += Aa;
  float* p1  = w; w += P;      // layer1 paper out
  float* fs  = w; w += P;      // src transform for current relation
  float* accb= w; w += P;      // unnormalized message accumulator
  float* elb = w; w += (size_t)Np * 4;
  float* erb = w; w += (size_t)Np * 4;
  float* sb  = w; w += (size_t)Np * 4;
  float* war = w; w += 1024;
  int*   mb  = (int*)w; w += (size_t)Np * 4;
  _Float16* hh = (_Float16*)w;
  _Float16* xp16 = hh; hh += (size_t)Np * 128;   // f16 mirrors of GEMM A operands
  _Float16* xa16 = hh; hh += (size_t)Na * 128;
  _Float16* p016 = hh; hh += P;
  _Float16* a016 = hh; hh += Aa;
  _Float16* p116 = hh;

  const int ENC_NEG_INF = (int)0x807FFFFFu;  // fenc(-inf)

  auto run_gat = [&](const _Float16* Xs16, int Ns, int Ks, const float* Xd, int Nd,
                     const float* W, const float* al, const float* ar, const float* b,
                     const int* src, const int* dst, int E, float* O) {
    dim3 gg(HD / 64, gblocks(Ns, 64));
    gemm_wmma_f16<<<gg, 256, 0, stream>>>(Xs16, W, fs, nullptr, Ns, HD, Ks);
    el_reduce<<<gblocks((long)Ns * 4, 256), 256, 0, stream>>>(fs, al, elb, Ns);
    w_ar_reduce<<<gblocks((long)Ks * 4, 64), 64, 0, stream>>>(W, ar, war, Ks);
    er_reduce<<<gblocks((long)Nd * 4, 256), 256, 0, stream>>>(Xd, war, erb, Nd, Ks);
    fill_i32<<<gblocks((long)Nd * 4, 256), 256, 0, stream>>>(mb, ENC_NEG_INF, (long)Nd * 4);
    fill_f32<<<gblocks((long)Nd * 4, 256), 256, 0, stream>>>(sb, 0.f, (long)Nd * 4);
    fill_f32<<<gblocks((long)Nd * HD, 256), 256, 0, stream>>>(accb, 0.f, (long)Nd * HD);
    edge_max<<<gblocks(E, 256), 256, 0, stream>>>(src, dst, elb, erb, mb, E);
    edge_acc<<<gblocks(E, 8), 256, 0, stream>>>(src, dst, elb, erb, mb, fs, sb, accb, E);
    finalize_acc<<<gblocks((long)Nd * HD, 256), 256, 0, stream>>>(accb, sb, b, O, Nd);
  };

  // f16 mirrors of the raw inputs
  cvt_f16<<<gblocks((long)Np * 128, 256), 256, 0, stream>>>(x_paper,  xp16, (long)Np * 128);
  cvt_f16<<<gblocks((long)Na * 128, 256), 256, 0, stream>>>(x_author, xa16, (long)Na * 128);

  // ===== layer 0 =====
  fill_f32<<<gblocks((long)P, 256), 256, 0, stream>>>(p0, 0.f, (long)P);
  fill_f32<<<gblocks((long)Aa, 256), 256, 0, stream>>>(a0, 0.f, (long)Aa);
  run_gat(xp16, Np, 128, x_paper,  Np, W0,             al0,       ar0,       b0,
          src_c,  dst_c,  Ec,  p0);
  run_gat(xa16, Na, 128, x_paper,  Np, W0 + 128*256,   al0 + 256, ar0 + 256, b0 + 256,
          src_w,  dst_w,  Ew,  p0);
  run_gat(xp16, Np, 128, x_author, Na, W0 + 2*128*256, al0 + 512, ar0 + 512, b0 + 512,
          src_wr, dst_wr, Ewr, a0);
  relu_cvt<<<gblocks((long)P, 256), 256, 0, stream>>>(p0, p016, (long)P);
  relu_cvt<<<gblocks((long)Aa, 256), 256, 0, stream>>>(a0, a016, (long)Aa);

  // ===== layer 1 (author output is dead: only paper feeds the final linear) =====
  fill_f32<<<gblocks((long)P, 256), 256, 0, stream>>>(p1, 0.f, (long)P);
  run_gat(p016, Np, 256, p0, Np, W1,           al1,       ar1,       b1,
          src_c, dst_c, Ec, p1);
  run_gat(a016, Na, 256, p0, Np, W1 + 256*256, al1 + 256, ar1 + 256, b1 + 256,
          src_w, dst_w, Ew, p1);

  // ===== final linear: d_out[Np,192] = p1 @ lw + lb =====
  cvt_f16<<<gblocks((long)P, 256), 256, 0, stream>>>(p1, p116, (long)P);
  dim3 gf(192 / 64, gblocks(Np, 64));
  gemm_wmma_f16<<<gf, 256, 0, stream>>>(p116, lw, (float*)d_out, lb, Np, 192, 256);
}
